// MetaLayerGNN_72954314490172
// MI455X (gfx1250) — matrix operand order repaired and checked by reading, not compiled, not run on hardware
//
#include <hip/hip_runtime.h>

typedef __attribute__((ext_vector_type(16))) _Float16 v16h;
typedef __attribute__((ext_vector_type(8)))  float    v8f;

#define WMMA(a, b, c) \
  __builtin_amdgcn_wmma_f32_16x16x32_f16(false, (a), false, (b), (short)0, (c), false, false)

// ---------- fragment helpers ----------
// A-fragment (16x32 f16, M x K): lane L<16 holds row L, K = {0..15};
// lane L>=16 holds row L-16, K = {16..31}.  c0 = (lane<16)?0:16.
__device__ __forceinline__ v16h a_frag_global(const float* p, int c0, float scale) {
  const float4* q0 = (const float4*)(p + c0);
  const float4* q1 = (const float4*)(p + c0 + 8);
  float4 f0 = q0[0], f1 = q0[1], f2 = q1[0], f3 = q1[1];
  v16h a;
  a[0]  = (_Float16)(f0.x * scale); a[1]  = (_Float16)(f0.y * scale);
  a[2]  = (_Float16)(f0.z * scale); a[3]  = (_Float16)(f0.w * scale);
  a[4]  = (_Float16)(f1.x * scale); a[5]  = (_Float16)(f1.y * scale);
  a[6]  = (_Float16)(f1.z * scale); a[7]  = (_Float16)(f1.w * scale);
  a[8]  = (_Float16)(f2.x * scale); a[9]  = (_Float16)(f2.y * scale);
  a[10] = (_Float16)(f2.z * scale); a[11] = (_Float16)(f2.w * scale);
  a[12] = (_Float16)(f3.x * scale); a[13] = (_Float16)(f3.y * scale);
  a[14] = (_Float16)(f3.z * scale); a[15] = (_Float16)(f3.w * scale);
  return a;
}

// B-fragment preswizzled in LDS: 16 contiguous halves per lane.
__device__ __forceinline__ v16h b_frag_lds(const _Float16* p) {
  v16h b;
#pragma unroll
  for (int i = 0; i < 16; ++i) b[i] = p[i];
  return b;
}

// A-fragment from row-major f16 LDS tile (hidden activations), row stride 64.
__device__ __forceinline__ v16h a_frag_h(const _Float16* p) {
  v16h a;
#pragma unroll
  for (int i = 0; i < 8; ++i) { a[i] = p[i]; a[8 + i] = p[8 + i]; }
  return a;
}

// B-fragment straight from global (used only by tiny global-MLP kernel).
__device__ __forceinline__ v16h b_frag_global(const float* W, int ld, int kk, int nn, int lane) {
  int kr = kk * 32 + ((lane & 16) ? 16 : 0);
  int nc = nn * 16 + (lane & 15);
  v16h b;
#pragma unroll
  for (int i = 0; i < 16; ++i) b[i] = (_Float16)W[(kr + i) * ld + nc];
  return b;
}

__device__ __forceinline__ void atomic_add_f32(float* p, float v) {
  __hip_atomic_fetch_add(p, v, __ATOMIC_RELAXED, __HIP_MEMORY_SCOPE_AGENT);
}

// ---------- utility kernels ----------
__global__ void zero_f32_kernel(float* p, int n) {
  for (int i = blockIdx.x * blockDim.x + threadIdx.x; i < n; i += gridDim.x * blockDim.x)
    p[i] = 0.0f;
}

__global__ void bn_kernel(const float* in, float* out, const float* gamma,
                          const float* beta, int n) {
  int i = blockIdx.x * blockDim.x + threadIdx.x;
  if (i < n) {
    int c = i & 31;
    out[i] = in[i] * (gamma[c] * rsqrtf(1.0f + 1e-5f)) + beta[c];
  }
}

__global__ void count_kernel(const int* idx, float* cnt, int n) {
  int i = blockIdx.x * blockDim.x + threadIdx.x;
  if (i < n) atomic_add_f32(cnt + idx[i], 1.0f);
}

// ---------- edge model: e' = MLP([x[src], x[dst], e, u[batch[src]]]) ----------
// 16 edges per wave; GEMM1: 16x128 @ 128x64 (4 K-chunks x 4 N-tiles WMMA),
// ReLU via LDS, GEMM2: 16x64 @ 64x32 (2x2), scatter-add e' into agg[dst].
// Weight B-fragments are staged in LDS once per block, then held in VGPRs
// across the whole tile loop (160 VGPRs) so no DS waits gate the WMMAs.
__global__ __launch_bounds__(256) void edge_kernel(
    const float* x, const float* e_in, float* e_out, const float* u,
    const int* src, const int* dst, const int* batch,
    const float* W1, const float* b1, const float* W2, const float* b2,
    float* agg, int E) {
  __shared__ _Float16 w1f[16 * 32 * 16];  // 16 frags (kk 0..3, nn 0..3)
  __shared__ _Float16 w2f[4 * 32 * 16];   // 4 frags (kk 0..1, nn 0..1)
  __shared__ float    b1s[64];
  __shared__ float    b2s[32];
  __shared__ _Float16 hbuf[8 * 16 * 64];  // per-wave hidden tile

  for (int idx = threadIdx.x; idx < 16 * 32; idx += blockDim.x) {
    int f = idx >> 5, L = idx & 31;
    int kk = f >> 2, nn = f & 3;
    int kr = kk * 32 + ((L & 16) ? 16 : 0);
    int nc = nn * 16 + (L & 15);
#pragma unroll
    for (int i = 0; i < 16; ++i) w1f[idx * 16 + i] = (_Float16)W1[(kr + i) * 64 + nc];
  }
  for (int idx = threadIdx.x; idx < 4 * 32; idx += blockDim.x) {
    int f = idx >> 5, L = idx & 31;
    int kk = f >> 1, nn = f & 1;
    int kr = kk * 32 + ((L & 16) ? 16 : 0);
    int nc = nn * 16 + (L & 15);
#pragma unroll
    for (int i = 0; i < 16; ++i) w2f[idx * 16 + i] = (_Float16)W2[(kr + i) * 32 + nc];
  }
  if (threadIdx.x < 64) b1s[threadIdx.x] = b1[threadIdx.x];
  if (threadIdx.x < 32) b2s[threadIdx.x] = b2[threadIdx.x];
  __syncthreads();

  const int lane = threadIdx.x & 31;
  const int wave = threadIdx.x >> 5;
  const int mh = (lane & 16) ? 8 : 0;   // C/D row offset
  const int c0 = (lane & 16) ? 16 : 0;  // A K offset
  const int ln = lane & 15;
  _Float16* hw = hbuf + wave * 16 * 64;

  // hoist weight fragments into registers (loop-invariant)
  v16h w1r[16];
#pragma unroll
  for (int f = 0; f < 16; ++f) w1r[f] = b_frag_lds(w1f + (f * 32 + lane) * 16);
  v16h w2r[4];
#pragma unroll
  for (int f = 0; f < 4; ++f) w2r[f] = b_frag_lds(w2f + (f * 32 + lane) * 16);
  const float bv1[4] = { b1s[ln], b1s[16 + ln], b1s[32 + ln], b1s[48 + ln] };
  const float bv2[2] = { b2s[ln], b2s[16 + ln] };

  const int ntiles = E >> 4;
  for (int t = blockIdx.x * 8 + wave; t < ntiles; t += gridDim.x * 8) {
    const int row = (t << 4) + ln;
    const int s = src[row];
    const int d = dst[row];
    const int bs = batch[s];

    v8f acc[4];
#pragma unroll
    for (int nn = 0; nn < 4; ++nn) {
#pragma unroll
      for (int j = 0; j < 8; ++j) acc[nn][j] = bv1[nn];
    }
    const float* segp[4] = { x + (size_t)s * 32, x + (size_t)d * 32,
                             e_in + (size_t)row * 32, u + (size_t)bs * 32 };
#pragma unroll
    for (int kk = 0; kk < 4; ++kk) {
      v16h a = a_frag_global(segp[kk], c0, 1.0f);
#pragma unroll
      for (int nn = 0; nn < 4; ++nn)
        acc[nn] = WMMA(a, w1r[kk * 4 + nn], acc[nn]);
    }
    // ReLU -> LDS (reshape D layout -> A layout)
#pragma unroll
    for (int nn = 0; nn < 4; ++nn) {
      int n = nn * 16 + ln;
#pragma unroll
      for (int j = 0; j < 8; ++j) {
        float v = acc[nn][j];
        hw[(j + mh) * 64 + n] = (_Float16)(v > 0.0f ? v : 0.0f);
      }
    }
    asm volatile("s_wait_dscnt 0" ::: "memory");

    v8f acc2[2];
#pragma unroll
    for (int nn = 0; nn < 2; ++nn) {
#pragma unroll
      for (int j = 0; j < 8; ++j) acc2[nn][j] = bv2[nn];
    }
#pragma unroll
    for (int kk = 0; kk < 2; ++kk) {
      v16h a = a_frag_h(hw + ln * 64 + kk * 32 + c0);
#pragma unroll
      for (int nn = 0; nn < 2; ++nn)
        acc2[nn] = WMMA(a, w2r[kk * 2 + nn], acc2[nn]);
    }
    // write e' and scatter-add into agg[dst]
#pragma unroll
    for (int nn = 0; nn < 2; ++nn) {
      int n = nn * 16 + ln;
#pragma unroll
      for (int j = 0; j < 8; ++j) {
        int r = (t << 4) + j + mh;
        float v = acc2[nn][j];
        e_out[(size_t)r * 32 + n] = v;
        atomic_add_f32(agg + (size_t)dst[r] * 32 + n, v);
      }
    }
  }
}

// ---------- node model: x' = MLP([x, agg/deg, u[batch]]) ----------
__global__ __launch_bounds__(256) void node_kernel(
    const float* x_in, float* x_out, const float* agg, const float* deg,
    const float* u, const int* batch,
    const float* W1, const float* b1, const float* W2, const float* b2,
    float* xg, int N) {
  __shared__ _Float16 w1f[12 * 32 * 16];  // 12 frags (kk 0..2, nn 0..3)
  __shared__ _Float16 w2f[4 * 32 * 16];
  __shared__ float    b1s[64];
  __shared__ float    b2s[32];
  __shared__ _Float16 hbuf[8 * 16 * 64];

  for (int idx = threadIdx.x; idx < 12 * 32; idx += blockDim.x) {
    int f = idx >> 5, L = idx & 31;
    int kk = f >> 2, nn = f & 3;
    int kr = kk * 32 + ((L & 16) ? 16 : 0);
    int nc = nn * 16 + (L & 15);
#pragma unroll
    for (int i = 0; i < 16; ++i) w1f[idx * 16 + i] = (_Float16)W1[(kr + i) * 64 + nc];
  }
  for (int idx = threadIdx.x; idx < 4 * 32; idx += blockDim.x) {
    int f = idx >> 5, L = idx & 31;
    int kk = f >> 1, nn = f & 1;
    int kr = kk * 32 + ((L & 16) ? 16 : 0);
    int nc = nn * 16 + (L & 15);
#pragma unroll
    for (int i = 0; i < 16; ++i) w2f[idx * 16 + i] = (_Float16)W2[(kr + i) * 32 + nc];
  }
  if (threadIdx.x < 64) b1s[threadIdx.x] = b1[threadIdx.x];
  if (threadIdx.x < 32) b2s[threadIdx.x] = b2[threadIdx.x];
  __syncthreads();

  const int lane = threadIdx.x & 31;
  const int wave = threadIdx.x >> 5;
  const int mh = (lane & 16) ? 8 : 0;
  const int c0 = (lane & 16) ? 16 : 0;
  const int ln = lane & 15;
  _Float16* hw = hbuf + wave * 16 * 64;

  v16h w1r[12];
#pragma unroll
  for (int f = 0; f < 12; ++f) w1r[f] = b_frag_lds(w1f + (f * 32 + lane) * 16);
  v16h w2r[4];
#pragma unroll
  for (int f = 0; f < 4; ++f) w2r[f] = b_frag_lds(w2f + (f * 32 + lane) * 16);
  const float bv1[4] = { b1s[ln], b1s[16 + ln], b1s[32 + ln], b1s[48 + ln] };
  const float bv2[2] = { b2s[ln], b2s[16 + ln] };

  const int ntiles = N >> 4;
  for (int t = blockIdx.x * 8 + wave; t < ntiles; t += gridDim.x * 8) {
    const int row = (t << 4) + ln;
    const int bb = batch[row];
    const float invd = 1.0f / fmaxf(deg[row], 1.0f);

    v8f acc[4];
#pragma unroll
    for (int nn = 0; nn < 4; ++nn) {
#pragma unroll
      for (int j = 0; j < 8; ++j) acc[nn][j] = bv1[nn];
    }
    const float* segp[3] = { x_in + (size_t)row * 32, agg + (size_t)row * 32,
                             u + (size_t)bb * 32 };
    const float segs[3] = { 1.0f, invd, 1.0f };
#pragma unroll
    for (int kk = 0; kk < 3; ++kk) {
      v16h a = a_frag_global(segp[kk], c0, segs[kk]);
#pragma unroll
      for (int nn = 0; nn < 4; ++nn)
        acc[nn] = WMMA(a, w1r[kk * 4 + nn], acc[nn]);
    }
#pragma unroll
    for (int nn = 0; nn < 4; ++nn) {
      int n = nn * 16 + ln;
#pragma unroll
      for (int j = 0; j < 8; ++j) {
        float v = acc[nn][j];
        hw[(j + mh) * 64 + n] = (_Float16)(v > 0.0f ? v : 0.0f);
      }
    }
    asm volatile("s_wait_dscnt 0" ::: "memory");

    v8f acc2[2];
#pragma unroll
    for (int nn = 0; nn < 2; ++nn) {
#pragma unroll
      for (int j = 0; j < 8; ++j) acc2[nn][j] = bv2[nn];
    }
#pragma unroll
    for (int kk = 0; kk < 2; ++kk) {
      v16h a = a_frag_h(hw + ln * 64 + kk * 32 + c0);
#pragma unroll
      for (int nn = 0; nn < 2; ++nn)
        acc2[nn] = WMMA(a, w2r[kk * 2 + nn], acc2[nn]);
    }
#pragma unroll
    for (int nn = 0; nn < 2; ++nn) {
      int n = nn * 16 + ln;
#pragma unroll
      for (int j = 0; j < 8; ++j) {
        int r = (t << 4) + j + mh;
        float v = acc2[nn][j];
        x_out[(size_t)r * 32 + n] = v;
        atomic_add_f32(xg + (size_t)batch[r] * 32 + n, v);
      }
    }
  }
}

// ---------- global model: u' = MLP([u, xg/gcnt]) ; B==16 -> single wave ----------
__global__ __launch_bounds__(32) void glob_kernel(
    const float* u_in, float* u_out, const float* xg, const float* gcnt,
    const float* W1, const float* b1, const float* W2, const float* b2) {
  __shared__ _Float16 hbuf[16 * 64];
  const int lane = threadIdx.x & 31;
  const int mh = (lane & 16) ? 8 : 0;
  const int c0 = (lane & 16) ? 16 : 0;
  const int ln = lane & 15;
  const int row = ln;
  const float invg = 1.0f / fmaxf(gcnt[row], 1.0f);

  v8f acc[4];
#pragma unroll
  for (int nn = 0; nn < 4; ++nn) {
    float bv = b1[nn * 16 + ln];
#pragma unroll
    for (int j = 0; j < 8; ++j) acc[nn][j] = bv;
  }
  const float* segp[2] = { u_in + (size_t)row * 32, xg + (size_t)row * 32 };
  const float segs[2] = { 1.0f, invg };
#pragma unroll
  for (int kk = 0; kk < 2; ++kk) {
    v16h a = a_frag_global(segp[kk], c0, segs[kk]);
#pragma unroll
    for (int nn = 0; nn < 4; ++nn)
      acc[nn] = WMMA(a, b_frag_global(W1, 64, kk, nn, lane), acc[nn]);
  }
#pragma unroll
  for (int nn = 0; nn < 4; ++nn) {
    int n = nn * 16 + ln;
#pragma unroll
    for (int j = 0; j < 8; ++j) {
      float v = acc[nn][j];
      hbuf[(j + mh) * 64 + n] = (_Float16)(v > 0.0f ? v : 0.0f);
    }
  }
  asm volatile("s_wait_dscnt 0" ::: "memory");

  v8f acc2[2];
#pragma unroll
  for (int nn = 0; nn < 2; ++nn) {
    float bv = b2[nn * 16 + ln];
#pragma unroll
    for (int j = 0; j < 8; ++j) acc2[nn][j] = bv;
  }
#pragma unroll
  for (int kk = 0; kk < 2; ++kk) {
    v16h a = a_frag_h(hbuf + ln * 64 + kk * 32 + c0);
#pragma unroll
    for (int nn = 0; nn < 2; ++nn)
      acc2[nn] = WMMA(a, b_frag_global(W2, 32, kk, nn, lane), acc2[nn]);
  }
#pragma unroll
  for (int nn = 0; nn < 2; ++nn) {
    int n = nn * 16 + ln;
#pragma unroll
    for (int j = 0; j < 8; ++j)
      u_out[(size_t)(j + mh) * 32 + n] = acc2[nn][j];
  }
}

// ---------- host launch ----------
extern "C" void kernel_launch(void* const* d_in, const int* in_sizes, int n_in,
                              void* d_out, int out_size, void* d_ws, size_t ws_size,
                              hipStream_t stream) {
  const int N = in_sizes[0] / 32;   // 50000
  const int E = in_sizes[2] / 32;   // 800000
  const int B = in_sizes[3] / 32;   // 16

  const float* node_feats = (const float*)d_in[0];
  const int*   edge_index = (const int*)d_in[1];
  const float* edge_feats = (const float*)d_in[2];
  const float* glob_feats = (const float*)d_in[3];
  const int*   batch      = (const int*)d_in[4];
  const float* node_gamma = (const float*)d_in[5];
  const float* node_beta  = (const float*)d_in[6];
  const float* edge_gamma = (const float*)d_in[7];
  const float* edge_beta  = (const float*)d_in[8];
  const float* glob_gamma = (const float*)d_in[9];
  const float* glob_beta  = (const float*)d_in[10];
  const float* eW1 = (const float*)d_in[11];
  const float* eb1 = (const float*)d_in[12];
  const float* eW2 = (const float*)d_in[13];
  const float* eb2 = (const float*)d_in[14];
  const float* nW1 = (const float*)d_in[15];
  const float* nb1 = (const float*)d_in[16];
  const float* nW2 = (const float*)d_in[17];
  const float* nb2 = (const float*)d_in[18];
  const float* gW1 = (const float*)d_in[19];
  const float* gb1 = (const float*)d_in[20];
  const float* gW2 = (const float*)d_in[21];
  const float* gb2 = (const float*)d_in[22];

  const int* src = edge_index;
  const int* dst = edge_index + E;

  float* ws   = (float*)d_ws;
  float* x    = ws;
  float* e    = x + (size_t)N * 32;
  float* u    = e + (size_t)E * 32;
  float* agg  = u + (size_t)B * 32;
  float* xg   = agg + (size_t)N * 32;   // contiguous with agg (zeroed together)
  float* deg  = xg + (size_t)B * 32;    // contiguous with gcnt (zeroed together)
  float* gcnt = deg + N;

  float* xout = (float*)d_out;
  float* eout = xout + (size_t)N * 32;
  float* uout = eout + (size_t)E * 32;

  zero_f32_kernel<<<256, 256, 0, stream>>>(deg, N + B);
  bn_kernel<<<(N * 32 + 255) / 256, 256, 0, stream>>>(node_feats, x, node_gamma, node_beta, N * 32);
  bn_kernel<<<(E * 32 + 255) / 256, 256, 0, stream>>>(edge_feats, e, edge_gamma, edge_beta, E * 32);
  bn_kernel<<<(B * 32 + 255) / 256, 256, 0, stream>>>(glob_feats, u, glob_gamma, glob_beta, B * 32);
  count_kernel<<<(E + 255) / 256, 256, 0, stream>>>(dst, deg, E);
  count_kernel<<<(N + 255) / 256, 256, 0, stream>>>(batch, gcnt, N);

  for (int l = 0; l < 3; ++l) {
    zero_f32_kernel<<<1024, 256, 0, stream>>>(agg, (N + B) * 32);
    edge_kernel<<<512, 256, 0, stream>>>(
        x, e, (l == 2) ? eout : e, u, src, dst, batch,
        eW1 + (size_t)l * 128 * 64, eb1 + l * 64,
        eW2 + (size_t)l * 64 * 32,  eb2 + l * 32, agg, E);
    node_kernel<<<256, 256, 0, stream>>>(
        x, (l == 2) ? xout : x, agg, deg, u, batch,
        nW1 + (size_t)l * 96 * 64, nb1 + l * 64,
        nW2 + (size_t)l * 64 * 32, nb2 + l * 32, xg, N);
    glob_kernel<<<1, 32, 0, stream>>>(
        u, (l == 2) ? uout : u, xg, gcnt,
        gW1 + (size_t)l * 64 * 64, gb1 + l * 64,
        gW2 + (size_t)l * 64 * 32, gb2 + l * 32);
  }
}
